// RALCrossAttnDecoder_20727512170558
// MI455X (gfx1250) — matrix operand order repaired and checked by reading, not compiled, not run on hardware
//
#include <hip/hip_runtime.h>
#include <hip/hip_bf16.h>

// ---------------------------------------------------------------------------
// Problem constants (match reference)
// ---------------------------------------------------------------------------
#define NN      100000
#define CC      256
#define EMB     256
#define OUTD    8
#define HID     256
#define HEADS   4
#define KK      20000
#define RR      16
#define EE      800000
#define RETR_IN 265            // EMB + OUT + 1
#define KV_KP   288            // RETR_IN padded to multiple of 32
#define M1      (KK * RR)      // 320000 rows for retrieval projections

// ---------------------------------------------------------------------------
// bf16 helpers (round-to-nearest-even)
// ---------------------------------------------------------------------------
__device__ __forceinline__ unsigned short f2bf(float f) {
  unsigned u = __float_as_uint(f);
  unsigned r = u + 0x7FFFu + ((u >> 16) & 1u);
  return (unsigned short)(r >> 16);
}
__device__ __forceinline__ float bf2f(unsigned short h) {
  return __uint_as_float(((unsigned)h) << 16);
}

typedef __attribute__((ext_vector_type(16))) __bf16           v16bf;
typedef __attribute__((ext_vector_type(8)))  float            v8f;
typedef __attribute__((ext_vector_type(8)))  unsigned short   u16x8;

union FragU { u16x8 h[2]; v16bf v; };

// ---------------------------------------------------------------------------
// Generic WMMA GEMM:  C[M,Nout] = act(A[M,Kp](bf16) @ B[Kp,Nout] + bias)
// A:  row-major bf16, leading dim lda (elements), streamed global->VGPR (b128).
// BT: bf16, row n holds column n of B (rows padded to multiple of 128).
//     The 128-column B panel for this block is async-copied to LDS ONCE
//     (global_load_async_to_lds_b128, ASYNCcnt) and shared by all 4 waves.
// Each wave computes a 16x128 tile: 1 A-fragment feeds 8 v_wmma; B fragments
// come from LDS via ds_load_b128 (rows padded -> conflict-free banks).
// Block = 128 threads = 4 waves covering 64 rows of M.
// ---------------------------------------------------------------------------
__global__ __launch_bounds__(128) void wmma_gemm_bf16(
    const unsigned short* __restrict__ A, int lda,
    const unsigned short* __restrict__ BT, int Kp,
    const float* __restrict__ bias,
    void* __restrict__ Cout, int ldc,
    int M, int Nout, int relu, int out_bf16)
{
  extern __shared__ unsigned short ldsb[];   // [128][Kp+8] bf16 panel
  const int pitch = Kp + 8;                  // +16B pad: lanes 4 banks apart
  const int wave = threadIdx.x >> 5;
  const int lane = threadIdx.x & 31;
  const int ncol0 = blockIdx.x * 128;
  const int hl  = lane >> 4;                 // lane half (0/1)
  const int sub = lane & 15;

  // ---- async-stage B panel (128 cols x Kp) into LDS, one row per thread ----
  {
    const int row = threadIdx.x;             // 128 threads -> 128 panel rows
    const unsigned short* gsrc = BT + (size_t)(ncol0 + row) * (size_t)Kp;
    unsigned lbase = (unsigned)(size_t)(ldsb + (size_t)row * pitch);
    for (int c = 0; c < Kp; c += 8) {
      unsigned long long ga = (unsigned long long)(gsrc + c);
      unsigned la = lbase + (unsigned)(c * 2);
      asm volatile("global_load_async_to_lds_b128 %0, %1, off"
                   :: "v"(la), "v"(ga) : "memory");
    }
    asm volatile("s_wait_asynccnt 0x0" ::: "memory");
  }
  __syncthreads();

  const int mrow = (blockIdx.y * 4 + wave) * 16;
  if (mrow >= M) return;                     // wave-uniform after barrier

  const unsigned short* arow = A + (size_t)(mrow + sub) * (size_t)lda;
  // Per-lane LDS B base: column sub, K offset hl*16
  const unsigned short* brow = ldsb + (size_t)sub * pitch + hl * 16;

  v8f acc[8];
#pragma unroll
  for (int t = 0; t < 8; ++t) acc[t] = (v8f){};

  for (int kb = 0; kb < Kp; kb += 32) {
    // A fragment: 16x32 bf16.  lanes 0-15 -> K {kb..kb+15},
    // lanes 16-31 -> K {kb+16..kb+31}  (matches B contiguous order)
    FragU a;
    a.h[0] = *(const u16x8*)(arow + kb + hl * 16);
    a.h[1] = *(const u16x8*)(arow + kb + hl * 16 + 8);

#pragma unroll
    for (int t = 0; t < 8; ++t) {
      // B fragment: 32x16 bf16 from LDS; 16 contiguous K at kb + hl*16.
      const unsigned short* bp = brow + (size_t)(t * 16) * (size_t)pitch + kb;
      FragU fb;
      fb.h[0] = *(const u16x8*)(bp);
      fb.h[1] = *(const u16x8*)(bp + 8);
      acc[t] = __builtin_amdgcn_wmma_f32_16x16x32_bf16(false, a.v, false, fb.v,
                                                       (short)0, acc[t], false, false);
    }
  }

  // C/D layout: lanes 0-15 -> N=lane, M = mrow + r; lanes 16-31 -> M = mrow+8+r
  const int mbase = mrow + hl * 8;
#pragma unroll
  for (int t = 0; t < 8; ++t) {
    int n = ncol0 + t * 16 + sub;
    if (n >= Nout) continue;
    float bv = bias ? bias[n] : 0.0f;
#pragma unroll
    for (int r = 0; r < 8; ++r) {
      float v = acc[t][r] + bv;
      if (relu) v = fmaxf(v, 0.0f);
      size_t off = (size_t)(mbase + r) * (size_t)ldc + (size_t)n;
      if (out_bf16) ((unsigned short*)Cout)[off] = f2bf(v);
      else          ((float*)Cout)[off] = v;
    }
  }
}

// ---------------------------------------------------------------------------
// Weight pack: BT[n*Kp + k] = bf16(W[k*fo_total + col_off + n]), zero padded.
// ---------------------------------------------------------------------------
__global__ void pack_weight(const float* __restrict__ W, int fi, int fo_total,
                            int col_off, int nsub, int Kp, int Npad,
                            unsigned short* __restrict__ BT)
{
  size_t idx = (size_t)blockIdx.x * blockDim.x + threadIdx.x;
  size_t total = (size_t)Npad * (size_t)Kp;
  if (idx >= total) return;
  int n = (int)(idx / Kp);
  int k = (int)(idx % Kp);
  float v = 0.0f;
  if (n < nsub && k < fi) v = W[(size_t)k * fo_total + col_off + n];
  BT[idx] = f2bf(v);
}

// ---------------------------------------------------------------------------
// kv_in pack: [M1, 288] bf16 = concat(z_retr, o_retr, retr_score, zeros)
// ---------------------------------------------------------------------------
__global__ void pack_kvin(const float* __restrict__ z, const float* __restrict__ o,
                          const float* __restrict__ score,
                          unsigned short* __restrict__ dst)
{
  size_t idx = (size_t)blockIdx.x * blockDim.x + threadIdx.x;
  if (idx >= (size_t)M1 * KV_KP) return;
  int row = (int)(idx / KV_KP);
  int col = (int)(idx % KV_KP);
  float v = 0.0f;
  if (col < EMB)            v = z[(size_t)row * EMB + col];
  else if (col < EMB + OUTD) v = o[(size_t)row * OUTD + (col - EMB)];
  else if (col == EMB + OUTD) v = score[row];
  dst[idx] = f2bf(v);
}

// ---------------------------------------------------------------------------
// Gather hq = h_cone[nids] -> f32 copy + bf16 into gcat[:, 0:256]
// ---------------------------------------------------------------------------
__global__ void gather_hq(const float* __restrict__ h, const int* __restrict__ nids,
                          float* __restrict__ hqf, unsigned short* __restrict__ gcat)
{
  size_t idx = (size_t)blockIdx.x * blockDim.x + threadIdx.x;
  if (idx >= (size_t)KK * HID) return;
  int k = (int)(idx / HID);
  int c = (int)(idx % HID);
  float v = h[(size_t)nids[k] * CC + c];
  hqf[idx] = v;
  gcat[(size_t)k * (2 * HID) + c] = f2bf(v);
}

// ---------------------------------------------------------------------------
// Cross attention: per block one outpin k, per wave one head (HEADS=4, R=16).
// Lane handles dims d=lane and d=lane+32 of DH=64.
// ---------------------------------------------------------------------------
__global__ __launch_bounds__(128) void attn_kernel(
    const unsigned short* __restrict__ qh,   // [K, HID] bf16
    const unsigned short* __restrict__ kh,   // [K*R, HID] bf16
    const unsigned short* __restrict__ vh,   // [K*R, HID] bf16
    unsigned short* __restrict__ ctx)        // [K, HID] bf16
{
  int k    = blockIdx.x;
  int head = threadIdx.x >> 5;
  int lane = threadIdx.x & 31;
  int cbase = head * 64;
  const float scale = 0.125f;  // 1/sqrt(DH)

  float q0 = bf2f(qh[(size_t)k * HID + cbase + lane]);
  float q1 = bf2f(qh[(size_t)k * HID + cbase + lane + 32]);

  float s[RR];
  float mx = -3.0e38f;
#pragma unroll
  for (int r = 0; r < RR; ++r) {
    size_t row = ((size_t)k * RR + r) * HID + cbase;
    float p = q0 * bf2f(kh[row + lane]) + q1 * bf2f(kh[row + lane + 32]);
#pragma unroll
    for (int off = 16; off >= 1; off >>= 1) p += __shfl_xor(p, off, 32);
    p *= scale;
    s[r] = p;
    mx = fmaxf(mx, p);
  }
  float sum = 0.0f;
#pragma unroll
  for (int r = 0; r < RR; ++r) { s[r] = __expf(s[r] - mx); sum += s[r]; }
  float inv = 1.0f / sum;
  float c0 = 0.0f, c1 = 0.0f;
#pragma unroll
  for (int r = 0; r < RR; ++r) {
    size_t row = ((size_t)k * RR + r) * HID + cbase;
    float w = s[r] * inv;
    c0 += w * bf2f(vh[row + lane]);
    c1 += w * bf2f(vh[row + lane + 32]);
  }
  ctx[(size_t)k * HID + cbase + lane]      = f2bf(c0);
  ctx[(size_t)k * HID + cbase + lane + 32] = f2bf(c1);
}

// ---------------------------------------------------------------------------
// h copy (f32 + bf16 mirror)
// ---------------------------------------------------------------------------
__global__ void copy_h(const float* __restrict__ src, float* __restrict__ hf,
                       unsigned short* __restrict__ hb, size_t n)
{
  size_t idx = (size_t)blockIdx.x * blockDim.x + threadIdx.x;
  if (idx >= n) return;
  float v = src[idx];
  hf[idx] = v;
  hb[idx] = f2bf(v);
}

// Scatter gated update back into h (f32 and bf16 mirrors)
__global__ void scatter_upd(const int* __restrict__ nids,
                            const float* __restrict__ hqf,
                            const float* __restrict__ gate_logit,
                            const float* __restrict__ upd,
                            float* __restrict__ hf,
                            unsigned short* __restrict__ hb)
{
  size_t idx = (size_t)blockIdx.x * blockDim.x + threadIdx.x;
  if (idx >= (size_t)KK * CC) return;
  int k = (int)(idx / CC);
  int c = (int)(idx % CC);
  float g = 1.0f / (1.0f + __expf(-gate_logit[idx]));
  float v = hqf[idx] + g * upd[idx];
  size_t off = (size_t)nids[k] * CC + c;
  hf[off] = v;
  hb[off] = f2bf(v);
}

__global__ void fill_f32(float* __restrict__ p, float v, size_t n) {
  size_t idx = (size_t)blockIdx.x * blockDim.x + threadIdx.x;
  if (idx < n) p[idx] = v;
}

__global__ void deg_edges(const int* __restrict__ dst, float* __restrict__ deg) {
  size_t e = (size_t)blockIdx.x * blockDim.x + threadIdx.x;
  if (e >= EE) return;
  atomicAdd(&deg[dst[e]], 1.0f);
}

// Edge aggregation: one block per edge, 256 threads over feature dim.
__global__ __launch_bounds__(256) void agg_edges(
    const int* __restrict__ src, const int* __restrict__ dst,
    const float* __restrict__ deg, const float* __restrict__ x,
    float* __restrict__ agg)
{
  int e = blockIdx.x;
  int c = threadIdx.x;
  int s = src[e], d = dst[e];
  float nrm = rsqrtf(deg[s]) * rsqrtf(deg[d]);
  atomicAdd(&agg[(size_t)d * HID + c], x[(size_t)s * HID + c] * nrm);
}

// Self loop + optional ReLU + bf16 convert
__global__ void agg_finalize(const float* __restrict__ agg, const float* __restrict__ x,
                             const float* __restrict__ deg, unsigned short* __restrict__ outb,
                             int relu)
{
  size_t idx = (size_t)blockIdx.x * blockDim.x + threadIdx.x;
  if (idx >= (size_t)NN * HID) return;
  int i = (int)(idx / HID);
  float v = agg[idx] + x[idx] / deg[i];
  if (relu) v = fmaxf(v, 0.0f);
  outb[idx] = f2bf(v);
}

__global__ void final_add(const float* __restrict__ teacher, const float* __restrict__ delta,
                          float* __restrict__ out)
{
  size_t idx = (size_t)blockIdx.x * blockDim.x + threadIdx.x;
  if (idx >= (size_t)NN * OUTD) return;
  out[idx] = teacher[idx] + delta[idx];
}

// ---------------------------------------------------------------------------
// Host orchestration
// ---------------------------------------------------------------------------
extern "C" void kernel_launch(void* const* d_in, const int* in_sizes, int n_in,
                              void* d_out, int out_size, void* d_ws, size_t ws_size,
                              hipStream_t stream)
{
  (void)in_sizes; (void)n_in; (void)out_size; (void)ws_size;

  const float* h_cone     = (const float*)d_in[0];
  const float* z_retr     = (const float*)d_in[1];
  const float* o_retr     = (const float*)d_in[2];
  const float* retr_score = (const float*)d_in[3];
  const float* teacher    = (const float*)d_in[4];
  const int*   nids       = (const int*)d_in[5];
  const int*   eidx       = (const int*)d_in[6];
  const float* rp_W1 = (const float*)d_in[7],  *rp_b1 = (const float*)d_in[8];
  const float* rp_W2 = (const float*)d_in[9],  *rp_b2 = (const float*)d_in[10];
  const float* qp_W  = (const float*)d_in[11], *qp_b  = (const float*)d_in[12];
  const float* ipW   = (const float*)d_in[13], *ipb   = (const float*)d_in[14];
  const float* opW   = (const float*)d_in[15], *opb   = (const float*)d_in[16];
  const float* gW1   = (const float*)d_in[17], *gb1   = (const float*)d_in[18];
  const float* gW2   = (const float*)d_in[19], *gb2   = (const float*)d_in[20];
  const float* uW1   = (const float*)d_in[21], *ub1   = (const float*)d_in[22];
  const float* uW2   = (const float*)d_in[23], *ub2   = (const float*)d_in[24];
  const float* nW1   = (const float*)d_in[25], *nb1   = (const float*)d_in[26];
  const float* nW2   = (const float*)d_in[27], *nb2   = (const float*)d_in[28];
  const float* hW1   = (const float*)d_in[29], *hb1   = (const float*)d_in[30];
  const float* hW2   = (const float*)d_in[31], *hb2   = (const float*)d_in[32];

  const int* esrc = eidx;
  const int* edst = eidx + EE;

  // ---- workspace bump allocator --------------------------------------------
  char* base = (char*)d_ws;
  size_t cur = 0;
  auto alloc = [&](size_t bytes) -> void* {
    void* p = base + cur;
    cur = (cur + bytes + 255) & ~(size_t)255;
    return p;
  };

  // Big reusable regions
  unsigned short* RA = (unsigned short*)alloc((size_t)M1 * KV_KP * 2); // kvin -> kmat -> agg(f32)
  unsigned short* RB = (unsigned short*)alloc((size_t)M1 * HID * 2);   // kvh  -> vmat -> x2(f32)
  unsigned short* RC = (unsigned short*)alloc((size_t)M1 * HID * 2);   // kv   -> x1(f32)

  unsigned short* q_bf   = (unsigned short*)alloc((size_t)KK * HID * 2);
  unsigned short* qh_bf  = (unsigned short*)alloc((size_t)KK * HID * 2);
  unsigned short* ctx_bf = (unsigned short*)alloc((size_t)KK * HID * 2);
  unsigned short* gcat   = (unsigned short*)alloc((size_t)KK * 2 * HID * 2); // [hq | retr_ctx]
  unsigned short* g1_bf  = (unsigned short*)alloc((size_t)KK * HID * 2);
  unsigned short* u1_bf  = (unsigned short*)alloc((size_t)KK * HID * 2);
  float* gate_f  = (float*)alloc((size_t)KK * CC * 4);
  float* upd_f   = (float*)alloc((size_t)KK * CC * 4);
  float* hq_f    = (float*)alloc((size_t)KK * CC * 4);
  float* h_f     = (float*)alloc((size_t)NN * CC * 4);
  unsigned short* F1 = (unsigned short*)alloc((size_t)NN * HID * 2); // h_bf16 -> d1
  unsigned short* F2 = (unsigned short*)alloc((size_t)NN * HID * 2); // h1 -> h2
  float* deg_f   = (float*)alloc((size_t)NN * 4);
  float* delta_f = (float*)alloc((size_t)NN * OUTD * 4);

  // Packed weights (bf16, transposed, rows padded to multiple of 128)
  auto allocW = [&](int nsub, int Kp) -> unsigned short* {
    int Npad = ((nsub + 127) / 128) * 128;
    return (unsigned short*)alloc((size_t)Npad * Kp * 2);
  };
  unsigned short* rpW1T = allocW(HID, KV_KP);
  unsigned short* rpW2T = allocW(HID, HID);
  unsigned short* qpT   = allocW(HID, HID);
  unsigned short* WqT   = allocW(HID, HID);
  unsigned short* WkT   = allocW(HID, HID);
  unsigned short* WvT   = allocW(HID, HID);
  unsigned short* opT   = allocW(HID, HID);
  unsigned short* gW1T  = allocW(HID, 2 * HID);
  unsigned short* gW2T  = allocW(CC, HID);
  unsigned short* uW1T  = allocW(CC, HID);
  unsigned short* uW2T  = allocW(CC, CC);
  unsigned short* nW1T  = allocW(HID, CC);
  unsigned short* nW2T  = allocW(HID, HID);
  unsigned short* hW1T  = allocW(HID, HID);
  unsigned short* hW2T  = allocW(OUTD, HID);   // Npad = 128

  auto packW = [&](const float* W, int fi, int fo_total, int col_off, int nsub,
                   int Kp, unsigned short* BT) {
    int Npad = ((nsub + 127) / 128) * 128;
    size_t total = (size_t)Npad * Kp;
    pack_weight<<<(unsigned)((total + 255) / 256), 256, 0, stream>>>(
        W, fi, fo_total, col_off, nsub, Kp, Npad, BT);
  };
  auto gemm = [&](const void* A, int lda, const unsigned short* BT, int Kp,
                  const float* bias, void* C, int ldc, int M, int Nout,
                  int relu, int out_bf16) {
    dim3 g((Nout + 127) / 128, (M + 63) / 64);
    size_t shmem = (size_t)128 * (Kp + 8) * 2;   // B panel in LDS
    wmma_gemm_bf16<<<g, 128, shmem, stream>>>((const unsigned short*)A, lda, BT, Kp,
                                              bias, C, ldc, M, Nout, relu, out_bf16);
  };

  // ---- pack weights --------------------------------------------------------
  packW(rp_W1, RETR_IN, HID, 0, HID, KV_KP, rpW1T);
  packW(rp_W2, HID, HID, 0, HID, HID, rpW2T);
  packW(qp_W,  CC,  HID, 0, HID, HID, qpT);
  packW(ipW, HID, 3 * HID, 0,       HID, HID, WqT);
  packW(ipW, HID, 3 * HID, HID,     HID, HID, WkT);
  packW(ipW, HID, 3 * HID, 2 * HID, HID, HID, WvT);
  packW(opW, HID, HID, 0, HID, HID, opT);
  packW(gW1, CC + HID, HID, 0, HID, 2 * HID, gW1T);
  packW(gW2, HID, CC, 0, CC, HID, gW2T);
  packW(uW1, HID, CC, 0, CC, HID, uW1T);
  packW(uW2, CC, CC, 0, CC, CC, uW2T);
  packW(nW1, CC, HID, 0, HID, CC, nW1T);
  packW(nW2, HID, HID, 0, HID, HID, nW2T);
  packW(hW1, HID, HID, 0, HID, HID, hW1T);
  packW(hW2, HID, OUTD, 0, OUTD, HID, hW2T);

  // ---- retrieval projection chain -----------------------------------------
  {
    size_t total = (size_t)M1 * KV_KP;
    pack_kvin<<<(unsigned)((total + 255) / 256), 256, 0, stream>>>(z_retr, o_retr, retr_score, RA);
  }
  gemm(RA, KV_KP, rpW1T, KV_KP, rp_b1, RB, HID, M1, HID, 1, 1);  // kvh = relu(kvin@W1+b1)
  gemm(RB, HID,   rpW2T, HID,   rp_b2, RC, HID, M1, HID, 0, 1);  // kv  = kvh@W2+b2
  gemm(RC, HID,   WkT,   HID,   ipb + HID,     RA, HID, M1, HID, 0, 1); // kmat
  gemm(RC, HID,   WvT,   HID,   ipb + 2 * HID, RB, HID, M1, HID, 0, 1); // vmat

  // ---- query side ----------------------------------------------------------
  {
    size_t total = (size_t)KK * HID;
    gather_hq<<<(unsigned)((total + 255) / 256), 256, 0, stream>>>(h_cone, nids, hq_f, gcat);
  }
  gemm(gcat, 2 * HID, qpT, HID, qp_b, q_bf,  HID, KK, HID, 0, 1); // q  = hq@qpW+b
  gemm(q_bf, HID,     WqT, HID, ipb,  qh_bf, HID, KK, HID, 0, 1); // qh = q@Wq+bq

  // ---- attention -----------------------------------------------------------
  attn_kernel<<<KK, 128, 0, stream>>>(qh_bf, RA, RB, ctx_bf);

  // ---- out-proj + gated fusion --------------------------------------------
  gemm(ctx_bf, HID, opT, HID, opb, gcat + HID, 2 * HID, KK, HID, 0, 1); // retr_ctx -> gcat[:,256:]
  gemm(gcat, 2 * HID, gW1T, 2 * HID, gb1, g1_bf, HID, KK, HID, 1, 1);   // g1 = relu(gcat@gW1+b)
  gemm(g1_bf, HID, gW2T, HID, gb2, gate_f, CC, KK, CC, 0, 0);           // gate logits (f32)
  gemm(gcat + HID, 2 * HID, uW1T, HID, ub1, u1_bf, CC, KK, CC, 1, 1);   // u1 = relu(rc@uW1+b)
  gemm(u1_bf, CC, uW2T, CC, ub2, upd_f, CC, KK, CC, 0, 0);              // update (f32)

  {
    size_t total = (size_t)NN * CC;
    copy_h<<<(unsigned)((total + 255) / 256), 256, 0, stream>>>(h_cone, h_f, F1, total);
  }
  {
    size_t total = (size_t)KK * CC;
    scatter_upd<<<(unsigned)((total + 255) / 256), 256, 0, stream>>>(nids, hq_f, gate_f, upd_f, h_f, F1);
  }

  // ---- GCN layer 1 ---------------------------------------------------------
  fill_f32<<<(NN + 255) / 256, 256, 0, stream>>>(deg_f, 1.0f, (size_t)NN); // self loop
  deg_edges<<<(EE + 255) / 256, 256, 0, stream>>>(edst, deg_f);

  float* x1 = (float*)RC;
  gemm(F1, HID, nW1T, CC, nb1, x1, HID, NN, HID, 0, 0);                  // x1 = h@gnnW1+b (f32)

  float* agg = (float*)RA;
  {
    size_t total = (size_t)NN * HID;
    fill_f32<<<(unsigned)((total + 255) / 256), 256, 0, stream>>>(agg, 0.0f, total);
    agg_edges<<<EE, 256, 0, stream>>>(esrc, edst, deg_f, x1, agg);
    agg_finalize<<<(unsigned)((total + 255) / 256), 256, 0, stream>>>(agg, x1, deg_f, F2, 1); // h1 (bf16)
  }

  // ---- GCN layer 2 ---------------------------------------------------------
  float* x2 = (float*)RB;
  gemm(F2, HID, nW2T, HID, nb2, x2, HID, NN, HID, 0, 0);                 // x2 = h1@gnnW2+b (f32)
  {
    size_t total = (size_t)NN * HID;
    fill_f32<<<(unsigned)((total + 255) / 256), 256, 0, stream>>>(agg, 0.0f, total);
    agg_edges<<<EE, 256, 0, stream>>>(esrc, edst, deg_f, x2, agg);
    agg_finalize<<<(unsigned)((total + 255) / 256), 256, 0, stream>>>(agg, x2, deg_f, F2, 0); // h2 (bf16)
  }

  // ---- head ----------------------------------------------------------------
  gemm(F2, HID, hW1T, HID, hb1, F1, HID, NN, HID, 1, 1);                 // d1 = relu(h2@hW1+b)
  gemm(F1, HID, hW2T, HID, hb2, delta_f, OUTD, NN, OUTD, 0, 0);          // delta (f32)

  {
    size_t total = (size_t)NN * OUTD;
    final_add<<<(unsigned)((total + 255) / 256), 256, 0, stream>>>(teacher, delta_f, (float*)d_out);
  }
}
